// StokenAttention_1580547975408
// MI455X (gfx1250) — matrix operand order repaired and checked by reading, not compiled, not run on hardware
//
#include <hip/hip_runtime.h>
#include <math.h>

// ---------------------------------------------------------------------------
// StokenAttention forward, MI455X (gfx1250, wave32, WMMA bf16 -> f32 accum)
// Bandwidth-bound (~330MB traffic, ~2.7 GFLOP); WMMA for all GEMM stages.
// B operands staged TRANSPOSED in LDS so each lane's fragment is two aligned
// 16-byte runs (ds_load_b128) instead of 16 scalar ds_load_u16.
// ---------------------------------------------------------------------------

typedef __attribute__((ext_vector_type(16))) __bf16 bf16x16;
typedef __attribute__((ext_vector_type(8)))  __bf16 bf16x8;
typedef __attribute__((ext_vector_type(8)))  float  f32x8;

#define B_  16
#define C_  384
#define C3_ 1152
#define N_  64   // super tokens (8x8)
#define NH_ 8
#define HD_ 48

// --- WMMA fragment loaders (wave32 layouts per CDNA5 ISA 7.12.2) -----------
// A (16xK tile), row-major [row][k], lda % 8 == 0, k0 % 8 == 0:
// lane L: row = row0 + (L&15); elems = k0+(L>>4)*8 + [0..8) and +16+[0..8)
__device__ __forceinline__ bf16x16 ldsA(const __bf16* s, int lda, int row0, int k0, int lane) {
  const __bf16* p = s + (row0 + (lane & 15)) * lda + k0 + ((lane >> 4) << 3);
  bf16x8 lo = *(const bf16x8*)p;
  bf16x8 hi = *(const bf16x8*)(p + 16);
  bf16x16 a;
#pragma unroll
  for (int i = 0; i < 8; ++i) { a[i] = lo[i]; a[8 + i] = hi[i]; }
  return a;
}
// B (Kx16 tile) staged TRANSPOSED [n][k], ldt % 8 == 0, k0 % 16 == 0:
// lane L: col n = n0 + (L&15); elems = k0+(L>>4)*8 + [0..8) and +16+[0..8)
__device__ __forceinline__ bf16x16 ldsBt(const __bf16* s, int ldt, int k0, int n0, int lane) {
  const __bf16* p = s + (n0 + (lane & 15)) * ldt + k0 + ((lane >> 4) << 3);
  bf16x8 lo = *(const bf16x8*)p;
  bf16x8 hi = *(const bf16x8*)(p + 16);
  bf16x16 b;
#pragma unroll
  for (int i = 0; i < 8; ++i) { b[i] = lo[i]; b[8 + i] = hi[i]; }
  return b;
}
__device__ __forceinline__ f32x8 wmma_bf16(bf16x16 a, bf16x16 b, f32x8 c) {
  return __builtin_amdgcn_wmma_f32_16x16x32_bf16(false, a, false, b, (short)0, c, false, false);
}

// --- K1: stf[b,c,n] = mean of 8x8 block ------------------------------------
__global__ void k_block_mean(const float* __restrict__ x, float* __restrict__ stf) {
  int bc = blockIdx.x;            // b*C + c
  int t  = threadIdx.x;           // 0..63 -> block id
  const float* xp = x + (size_t)bc * 4096;
  int bi = t >> 3, bj = t & 7;
  float s = 0.f;
  for (int r = 0; r < 8; ++r) {
    const float* row = xp + (bi * 8 + r) * 64 + bj * 8;
#pragma unroll
    for (int q = 0; q < 8; ++q) s += row[q];
  }
  stf[(size_t)bc * 64 + t] = s * (1.f / 64.f);
}

// --- K2: affinity = softmax_k(pix @ s9 * scale); colsums -> aff_sum --------
#define PIXLD 392
#define S9LD  392
__global__ void k_affinity(const float* __restrict__ x, const float* __restrict__ stf,
                           float* __restrict__ aff, float* __restrict__ aff_sum) {
  __shared__ __align__(16) __bf16 pixA[64 * PIXLD];  // [p][c]
  __shared__ __align__(16) __bf16 s9t[16 * S9LD];    // [k][c]  (B transposed)
  __shared__ float  csum[128];
  int bn = blockIdx.x;
  int b = bn >> 6, n = bn & 63;
  int ni = n >> 3, nj = n & 7;
  int tid = threadIdx.x, lane = tid & 31, w = tid >> 5;

  for (int idx = tid; idx < 64 * 384; idx += 128) {
    int c = idx >> 6, p = idx & 63;
    float v = x[(((size_t)b * C_ + c) * 64 + ni * 8 + (p >> 3)) * 64 + nj * 8 + (p & 7)];
    pixA[p * PIXLD + c] = (__bf16)v;
  }
  for (int idx = tid; idx < 16 * 384; idx += 128) {
    int k = idx / 384, c = idx % 384;
    float v = 0.f;
    if (k < 9) {
      int yi = ni + k / 3 - 1, xj = nj + k % 3 - 1;
      if (yi >= 0 && yi < 8 && xj >= 0 && xj < 8)
        v = stf[((size_t)b * C_ + c) * 64 + yi * 8 + xj];
    }
    s9t[k * S9LD + c] = (__bf16)v;
  }
  __syncthreads();

  f32x8 acc = {};
  for (int kk = 0; kk < 384; kk += 32) {
    bf16x16 a  = ldsA(pixA, PIXLD, 16 * w, kk, lane);
    bf16x16 bb = ldsBt(s9t, S9LD, kk, 0, lane);
    acc = wmma_bf16(a, bb, acc);
  }

  const float scale = rsqrtf((float)C_);
  int col = lane & 15;
  float cs = 0.f;
  float* affp = aff + (size_t)bn * 64 * 16;
#pragma unroll
  for (int r = 0; r < 8; ++r) {
    int p = 16 * w + r + ((lane >> 4) << 3);
    float v = acc[r] * scale;
    float mx = (col < 9) ? v : -1e30f;
#pragma unroll
    for (int d = 1; d < 16; d <<= 1) mx = fmaxf(mx, __shfl_xor(mx, d, 32));
    float e = (col < 9) ? __expf(v - mx) : 0.f;
    float ss = e;
#pragma unroll
    for (int d = 1; d < 16; d <<= 1) ss += __shfl_xor(ss, d, 32);
    float av = e / ss;
    affp[p * 16 + col] = av;
    cs += av;
  }
  csum[tid] = cs;
  __syncthreads();
  if (tid < 9) {
    float tot = 0.f;
#pragma unroll
    for (int j = 0; j < 8; ++j) tot += csum[j * 16 + tid];
    int yi = ni + tid / 3 - 1, xj = nj + tid % 3 - 1;
    if (yi >= 0 && yi < 8 && xj >= 0 && xj < 8)
      atomicAdd(&aff_sum[b * 64 + yi * 8 + xj], tot);
  }
}

// --- K3: s = pix^T @ affinity, fold9-scatter into stf2 ---------------------
#define PTLD 72
#define AFLD 72
__global__ void k_aggregate(const float* __restrict__ x, const float* __restrict__ aff,
                            float* __restrict__ stf2) {
  __shared__ __align__(16) __bf16 pixT[384 * PTLD];  // [c][p]
  __shared__ __align__(16) __bf16 affBt[16 * AFLD];  // [k][p] (B transposed)
  int bn = blockIdx.x;
  int b = bn >> 6, n = bn & 63;
  int ni = n >> 3, nj = n & 7;
  int tid = threadIdx.x, lane = tid & 31, w = tid >> 5;

  for (int idx = tid; idx < 64 * 384; idx += 128) {
    int c = idx >> 6, p = idx & 63;
    float v = x[(((size_t)b * C_ + c) * 64 + ni * 8 + (p >> 3)) * 64 + nj * 8 + (p & 7)];
    pixT[c * PTLD + p] = (__bf16)v;
  }
  const float* affp = aff + (size_t)bn * 64 * 16;
  for (int idx = tid; idx < 16 * 64; idx += 128) {
    int k = idx >> 6, p = idx & 63;
    affBt[k * AFLD + p] = (__bf16)affp[p * 16 + k];
  }
  __syncthreads();

  int col = lane & 15;
  int yi = ni + col / 3 - 1, xj = nj + col % 3 - 1;
  bool valid = (col < 9) && yi >= 0 && yi < 8 && xj >= 0 && xj < 8;
  int nbr = yi * 8 + xj;
  for (int t = w; t < 24; t += 4) {     // 24 c-rowtiles of 16
    f32x8 acc = {};
    for (int kk = 0; kk < 64; kk += 32) {
      bf16x16 a  = ldsA(pixT, PTLD, 16 * t, kk, lane);
      bf16x16 bb = ldsBt(affBt, AFLD, kk, 0, lane);
      acc = wmma_bf16(a, bb, acc);
    }
    if (valid) {
#pragma unroll
      for (int r = 0; r < 8; ++r) {
        int c = 16 * t + r + ((lane >> 4) << 3);
        atomicAdd(&stf2[((size_t)b * C_ + c) * 64 + nbr], acc[r]);
      }
    }
  }
}

// --- K4: stf2 /= (aff_sum + 1e-12) -----------------------------------------
__global__ void k_normalize(float* __restrict__ stf2, const float* __restrict__ aff_sum) {
  int idx = blockIdx.x * blockDim.x + threadIdx.x;
  if (idx >= B_ * C_ * 64) return;
  int b = idx / (C_ * 64);
  int m = idx & 63;
  stf2[idx] /= (aff_sum[b * 64 + m] + 1e-12f);
}

// --- K5/K7: Y[b,o,m] = sum_c W[o,c] X[b,c,m] (+bias) ; K=384, N=64 ---------
#define ALD 136
#define XLD 136
__global__ void k_gemm64(const float* __restrict__ Wt, const float* __restrict__ X,
                         const float* __restrict__ bias, float* __restrict__ Y,
                         int Mrows, int rowBlocks) {
  __shared__ __align__(16) __bf16 As[64 * ALD];   // [o_local][k_chunk]
  __shared__ __align__(16) __bf16 Bst[64 * XLD];  // [m][k_chunk] (B transposed)
  int bx = blockIdx.x;
  int b = bx / rowBlocks;
  int O0 = (bx % rowBlocks) * 64;
  int tid = threadIdx.x, lane = tid & 31, w = tid >> 5;
  f32x8 acc[4] = {};
  for (int kc = 0; kc < 384; kc += 128) {
    for (int idx = tid; idx < 64 * 128; idx += 128) {
      int o = idx >> 7, k = idx & 127;
      As[o * ALD + k] = (__bf16)Wt[(size_t)(O0 + o) * 384 + kc + k];
    }
    for (int idx = tid; idx < 128 * 64; idx += 128) {
      int k = idx >> 6, m = idx & 63;
      Bst[m * XLD + k] = (__bf16)X[((size_t)b * 384 + kc + k) * 64 + m];
    }
    __syncthreads();
    for (int k2 = 0; k2 < 128; k2 += 32) {
      bf16x16 a = ldsA(As, ALD, 16 * w, k2, lane);
#pragma unroll
      for (int ct = 0; ct < 4; ++ct) {
        bf16x16 bb = ldsBt(Bst, XLD, k2, 16 * ct, lane);
        acc[ct] = wmma_bf16(a, bb, acc[ct]);
      }
    }
    __syncthreads();
  }
#pragma unroll
  for (int ct = 0; ct < 4; ++ct) {
#pragma unroll
    for (int r = 0; r < 8; ++r) {
      int orow = O0 + 16 * w + r + ((lane >> 4) << 3);
      int m = 16 * ct + (lane & 15);
      float v = acc[ct][r];
      if (bias) v += bias[orow];
      Y[((size_t)b * Mrows + orow) * 64 + m] = v;
    }
  }
}

// --- K6: per (b,h): S = k^T q * hd^-0.5; column softmax (axis n); o = v@attn
#define KTL 72
__global__ void k_attention(const float* __restrict__ qkv, float* __restrict__ obuf) {
  __shared__ __align__(16) __bf16 kT[64 * KTL];      // [n][d], d padded to 64
  __shared__ __align__(16) __bf16 qBt[64 * KTL];     // [m][d] (B transposed)
  __shared__ __align__(16) __bf16 vA[48 * KTL];      // [d][n]
  __shared__ float  Sf[64 * 65];                     // [n][m]
  __shared__ __align__(16) __bf16 attnBt[64 * KTL];  // [m][n] (B transposed)
  int bh = blockIdx.x;
  int b = bh >> 3, h = bh & 7;
  const float* base = qkv + ((size_t)b * C3_ + h * 144) * 64;
  const float* qp = base;
  const float* kp = base + 48 * 64;
  const float* vp = base + 96 * 64;
  int tid = threadIdx.x, lane = tid & 31, w = tid >> 5;

  for (int idx = tid; idx < 64 * 64; idx += 128) {
    int d = idx >> 6, m = idx & 63;      // global reads contiguous in m
    kT[m * KTL + d]  = (d < 48) ? (__bf16)kp[d * 64 + m] : (__bf16)0.f;
    qBt[m * KTL + d] = (d < 48) ? (__bf16)qp[d * 64 + m] : (__bf16)0.f;
  }
  for (int idx = tid; idx < 48 * 64; idx += 128) {
    int d = idx >> 6, n2 = idx & 63;
    vA[d * KTL + n2] = (__bf16)vp[d * 64 + n2];
  }
  __syncthreads();

  const float sc = rsqrtf((float)HD_);
  for (int mt = 0; mt < 4; ++mt) {     // wave w -> n-rowtile w
    f32x8 acc = {};
    for (int kk = 0; kk < 64; kk += 32) {
      bf16x16 a  = ldsA(kT, KTL, 16 * w, kk, lane);
      bf16x16 bb = ldsBt(qBt, KTL, kk, 16 * mt, lane);
      acc = wmma_bf16(a, bb, acc);
    }
#pragma unroll
    for (int r = 0; r < 8; ++r) {
      int n2 = 16 * w + r + ((lane >> 4) << 3);
      int m  = 16 * mt + (lane & 15);
      Sf[n2 * 65 + m] = acc[r] * sc;
    }
  }
  __syncthreads();
  if (tid < 64) {                       // softmax down columns (axis n)
    int m = tid;
    float mx = -1e30f;
    for (int n2 = 0; n2 < 64; ++n2) mx = fmaxf(mx, Sf[n2 * 65 + m]);
    float s = 0.f;
    for (int n2 = 0; n2 < 64; ++n2) s += __expf(Sf[n2 * 65 + m] - mx);
    float inv = 1.f / s;
    for (int n2 = 0; n2 < 64; ++n2)
      attnBt[m * KTL + n2] = (__bf16)(__expf(Sf[n2 * 65 + m] - mx) * inv);
  }
  __syncthreads();
  for (int t = w; t < 12; t += 4) {     // 3 d-rowtiles x 4 m-coltiles
    int dt = t >> 2, mt = t & 3;
    f32x8 acc = {};
    for (int kk = 0; kk < 64; kk += 32) {
      bf16x16 a  = ldsA(vA, KTL, 16 * dt, kk, lane);
      bf16x16 bb = ldsBt(attnBt, KTL, kk, 16 * mt, lane);
      acc = wmma_bf16(a, bb, acc);
    }
#pragma unroll
    for (int r = 0; r < 8; ++r) {
      int d = 16 * dt + r + ((lane >> 4) << 3);
      int m = 16 * mt + (lane & 15);
      obuf[((size_t)b * C_ + h * 48 + d) * 64 + m] = acc[r];
    }
  }
}

// --- K8: pix_out[p,c] = sum_k aff[p,k] * s9_3[c,k]; scatter to pixel layout
#define S9TLD 40
#define RESLD 129
__global__ void k_output(const float* __restrict__ stf3, const float* __restrict__ aff,
                         float* __restrict__ out) {
  __shared__ __align__(16) __bf16 Aaff[64 * 32];     // [p][k] (k padded to 32)
  __shared__ __align__(16) __bf16 Bs9t[128 * S9TLD]; // [c_local][k] (B transposed)
  __shared__ float  res[64 * RESLD];                 // [p][c_local]
  int bn = blockIdx.x;
  int b = bn >> 6, n = bn & 63;
  int ni = n >> 3, nj = n & 7;
  int tid = threadIdx.x, lane = tid & 31, w = tid >> 5;
  const float* affp = aff + (size_t)bn * 64 * 16;

  for (int idx = tid; idx < 64 * 32; idx += 128) {
    int p = idx >> 5, k = idx & 31;
    Aaff[idx] = (k < 16) ? (__bf16)affp[p * 16 + k] : (__bf16)0.f;
  }
  for (int cc = 0; cc < 3; ++cc) {      // 3 chunks of 128 channels
    int c0 = cc * 128;
    for (int idx = tid; idx < 128 * 32; idx += 128) {
      int cl = idx & 127, k = idx >> 7;
      float v = 0.f;
      if (k < 9) {
        int yi = ni + k / 3 - 1, xj = nj + k % 3 - 1;
        if (yi >= 0 && yi < 8 && xj >= 0 && xj < 8)
          v = stf3[((size_t)b * C_ + c0 + cl) * 64 + yi * 8 + xj];
      }
      Bs9t[cl * S9TLD + k] = (__bf16)v;
    }
    __syncthreads();
    for (int t = w; t < 32; t += 4) {   // 4 p-rowtiles x 8 c-coltiles
      int pt = t >> 3, ct = t & 7;
      bf16x16 a  = ldsA(Aaff, 32, 16 * pt, 0, lane);
      bf16x16 bb = ldsBt(Bs9t, S9TLD, 0, 16 * ct, lane);
      f32x8 acc = {};
      acc = wmma_bf16(a, bb, acc);
#pragma unroll
      for (int r = 0; r < 8; ++r) {
        int p  = 16 * pt + r + ((lane >> 4) << 3);
        int cl = 16 * ct + (lane & 15);
        res[p * RESLD + cl] = acc[r];
      }
    }
    __syncthreads();
    for (int idx = tid; idx < 128 * 64; idx += 128) {
      int cl = idx >> 6, p = idx & 63;
      out[((size_t)b * C_ + c0 + cl) * 4096 + (ni * 8 + (p >> 3)) * 64 + nj * 8 + (p & 7)]
          = res[p * RESLD + cl];
    }
    __syncthreads();
  }
}

// ---------------------------------------------------------------------------
extern "C" void kernel_launch(void* const* d_in, const int* in_sizes, int n_in,
                              void* d_out, int out_size, void* d_ws, size_t ws_size,
                              hipStream_t stream) {
  (void)in_sizes; (void)n_in; (void)out_size; (void)ws_size;
  const float* x      = (const float*)d_in[0];
  const float* qkv_w  = (const float*)d_in[1];
  const float* proj_w = (const float*)d_in[2];
  const float* proj_b = (const float*)d_in[3];
  float* out = (float*)d_out;

  float* ws   = (float*)d_ws;
  float* stf  = ws;                                  // B*C*64
  float* aff  = stf  + (size_t)B_ * C_ * 64;         // B*N*64*16 (k padded)
  float* affs = aff  + (size_t)B_ * N_ * 64 * 16;    // B*64
  float* stf2 = affs + (size_t)B_ * 64;              // B*C*64
  float* qkvb = stf2 + (size_t)B_ * C_ * 64;         // B*1152*64
  float* obuf = qkvb + (size_t)B_ * C3_ * 64;        // B*C*64
  float* stf3 = obuf + (size_t)B_ * C_ * 64;         // B*C*64

  // zero atomic-accumulated buffers (affs + stf2 are contiguous)
  hipMemsetAsync(affs, 0, ((size_t)B_ * 64 + (size_t)B_ * C_ * 64) * sizeof(float), stream);

  k_block_mean<<<B_ * C_, 64, 0, stream>>>(x, stf);
  k_affinity <<<B_ * N_, 128, 0, stream>>>(x, stf, aff, affs);
  k_aggregate<<<B_ * N_, 128, 0, stream>>>(x, aff, stf2);
  k_normalize<<<(B_ * C_ * 64 + 255) / 256, 256, 0, stream>>>(stf2, affs);
  k_gemm64   <<<B_ * 18, 128, 0, stream>>>(qkv_w, stf2, nullptr, qkvb, C3_, 18);
  k_attention<<<B_ * NH_, 128, 0, stream>>>(qkvb, obuf);
  k_gemm64   <<<B_ * 6, 128, 0, stream>>>(proj_w, obuf, proj_b, stf3, C_, 6);
  k_output   <<<B_ * N_, 128, 0, stream>>>(stf3, aff, out);
}
